// ACS_31147102830962
// MI455X (gfx1250) — matrix operand order repaired and checked by reading, not compiled, not run on hardware
//
#include <hip/hip_runtime.h>
#include <hip/hip_bf16.h>

typedef unsigned short u16;
typedef unsigned int   u32;
typedef unsigned long long u64;
typedef __attribute__((ext_vector_type(16))) __bf16 v16bf;
typedef __attribute__((ext_vector_type(8)))  float  v8f;
typedef __attribute__((ext_vector_type(4)))  unsigned int u32x4;
typedef __attribute__((ext_vector_type(8)))  int i32x8;
typedef __attribute__((ext_vector_type(4)))  int i32x4;

#define HH   56
#define WW   56
#define HWSZ 3136            // 56*56
#define NB   16              // batch
#define NPIX 50176           // 16*3136
#define MOUT 256             // every conv outputs 256 channels
#define CIN0 512

__device__ __forceinline__ u16 f2bf(float f) {
    u32 u = __float_as_uint(f);
    u32 r = (u + 0x7FFFu + ((u >> 16) & 1u)) >> 16;   // RNE
    return (u16)r;
}

// ---------------------------------------------------------------------------
// BN fold prep: s = g*rsqrt(v+eps), b = beta - mean*s, for 4 BN param packs.
// bnbuf layout (floats): s31@0 b31@256 s33@512 b33@768 sa1@1024 ba1@1280 s2@1536 b2@1792
// ---------------------------------------------------------------------------
__global__ void bnprep_kernel(const float* __restrict__ bn31, const float* __restrict__ bn33,
                              const float* __restrict__ bna1, const float* __restrict__ bna2,
                              float* __restrict__ o) {
    int c = threadIdx.x;
    if (c >= 256) return;
    const float eps = 1e-5f;
    {
        float s = bn31[c] * rsqrtf(bn31[768 + c] + eps);
        o[c] = s; o[256 + c] = bn31[256 + c] - bn31[512 + c] * s;
    }
    {
        float s = bn33[c] * rsqrtf(bn33[768 + c] + eps);
        o[512 + c] = s; o[768 + c] = bn33[256 + c] - bn33[512 + c] * s;
    }
    {
        float s = bna1[c] * rsqrtf(bna1[768 + c] + eps);
        o[1024 + c] = s; o[1280 + c] = bna1[256 + c] - bna1[512 + c] * s;
    }
    {
        float s = bna2[c] * rsqrtf(bna2[768 + c] + eps);
        o[1536 + c] = s; o[1792 + c] = bna2[256 + c] - bna2[512 + c] * s;
    }
}

// ---------------------------------------------------------------------------
// Stable top-k(512) of 1024 scores (descending, ties -> lower index first).
// ---------------------------------------------------------------------------
__global__ void topk_kernel(const float* __restrict__ score,
                            int* __restrict__ ind, float* __restrict__ scale) {
    __shared__ float sc[1024];
    int i = threadIdx.x;
    sc[i] = score[i];
    __syncthreads();
    float mine = sc[i];
    int rank = 0;
    for (int j = 0; j < 1024; ++j) {
        float v = sc[j];
        rank += (v > mine) || (v == mine && j < i);
    }
    if (rank < 512) {
        ind[rank]   = i;
        scale[rank] = 1.0f / (1.0f + __expf(-mine));
    }
}

// ---------------------------------------------------------------------------
// x: fp32 NCHW [16,512,56,56] -> xb: bf16 "NHWC" [NPIX][512] via LDS transpose.
// ---------------------------------------------------------------------------
__global__ void xpose_kernel(const float* __restrict__ x, u16* __restrict__ xb) {
    __shared__ float tile[32][33];
    int t  = threadIdx.x;
    int tx = t & 31;
    int ty = t >> 5;
    int P0 = blockIdx.x * 32;
    int c0 = blockIdx.y * 32;
    int n  = P0 / HWSZ;
    int hw0 = P0 - n * HWSZ;
    #pragma unroll
    for (int r = 0; r < 4; ++r) {
        int cl = ty + r * 8;
        tile[cl][tx] = x[((size_t)n * CIN0 + c0 + cl) * HWSZ + hw0 + tx];
    }
    __syncthreads();
    #pragma unroll
    for (int r = 0; r < 4; ++r) {
        int hwl = ty + r * 8;
        xb[(size_t)(P0 + hwl) * CIN0 + c0 + tx] = f2bf(tile[tx][hwl]);
    }
}

// ---------------------------------------------------------------------------
// Weight packers -> bf16, layout A[m][kpos*Cin + cin], fold scalar fuse weight
// (fwIdx<0 -> 1.0) and optional per-out-channel scale (BN gamma/sqrt(var)).
// ---------------------------------------------------------------------------
__global__ void pack3_kernel(const float* __restrict__ w, u16* __restrict__ o, int Cin,
                             const float* __restrict__ fw, int fwIdx) {
    int idx = blockIdx.x * blockDim.x + threadIdx.x;
    int tot = MOUT * Cin * 9;
    if (idx >= tot) return;
    int m    = idx / (Cin * 9);
    int rem  = idx - m * (Cin * 9);
    int cin  = rem / 9;
    int kpos = rem - cin * 9;
    float mul = (fwIdx >= 0) ? fw[fwIdx] : 1.0f;
    o[(size_t)m * (9 * Cin) + kpos * Cin + cin] = f2bf(w[idx] * mul);
}

__global__ void pack1_kernel(const float* __restrict__ w, u16* __restrict__ o, int Cin,
                             const float* __restrict__ fw, int fwIdx,
                             const float* __restrict__ perM) {
    int idx = blockIdx.x * blockDim.x + threadIdx.x;
    int tot = MOUT * Cin;
    if (idx >= tot) return;
    int m = idx / Cin;
    float mul = (fwIdx >= 0) ? fw[fwIdx] : 1.0f;
    if (perM) mul *= perM[m];
    o[idx] = f2bf(w[idx] * mul);
}

// ---------------------------------------------------------------------------
// Implicit-GEMM conv via v_wmma_f32_16x16x32_bf16.
// Block tile M=128 x N=128 pixels, K-step 64; 8 waves as 2(M) x 4(N), each
// wave 64x32 => acc[4][2], 16 WMMA per K-step, one barrier per step.
// A (weight) tiles: Tensor Data Mover DMA straight into padded LDS rows
//   (pad_interval = 32 DWORDs = one 64-half row, pad_amount = 4 DWORDs -> LDR 72),
//   issued by wave 0, double-buffered, fenced with s_wait_tensorcnt + barrier.
// B (activation) tiles: register pipeline (needs per-pixel zero-padding).
// EPI 0: fp32 NCHW (+optional per-channel scale/bias); EPI 1: bf16 NHWC.
// ---------------------------------------------------------------------------
union Frag { v16bf v; uint4 q[2]; };
union Pk8  { u16 s[8]; uint4 q; };

template <int EPI>
__global__ __launch_bounds__(256)
void conv_wmma_kernel(const u16* __restrict__ wA, const u16* __restrict__ xIn,
                      int Cin, int ksize,
                      const float* __restrict__ sc, const float* __restrict__ bi,
                      float* __restrict__ outF, int chanBase, int outCtot,
                      u16* __restrict__ outBf) {
    constexpr int LDR = 72;                         // padded row (halves): 64 + 8
    __shared__ __align__(16) u16 As[2][128 * LDR];  // 2 x 128 x 64 bf16 (TDM-filled)
    __shared__ __align__(16) u16 Bs[2][128 * LDR];

    const int t      = threadIdx.x;
    const int p0     = blockIdx.x * 128;            // pixel tile base (NPIX%128==0)
    const int mBlock = blockIdx.y * 128;            // out-channel tile base
    const int Kw     = ksize * ksize * Cin;         // packed weight row length
    const int spp    = Cin >> 6;                    // K-steps per kernel position
    const int S      = ksize * ksize * spp;         // total K-steps
    const int hk     = ksize >> 1;

    // B loader mapping: 128 rows x 64 halves, 64B (4 x uint4) per thread
    const int lrow = t >> 1;
    const int lch  = (t & 1) * 32;

    // B-side pixel geometry (fixed per thread)
    const int p     = p0 + lrow;
    const int nimgB = p / HWSZ;
    const int hwB   = p - nimgB * HWSZ;
    const int hB    = hwB / WW;
    const int wB    = hwB - hB * WW;

    // wave / lane decomposition
    const int lane   = t & 31;
    const int half   = lane >> 4;
    const int rr     = lane & 15;
    const int waveId = t >> 5;
    const int wm0    = (waveId & 1) * 64;           // 2 M-waves x 64
    const int wn0    = (waveId >> 1) * 32;          // 4 N-waves x 32
    const bool tdmWave = (waveId == 0);

    v8f acc[4][2];
    #pragma unroll
    for (int i = 0; i < 4; ++i)
        #pragma unroll
        for (int j = 0; j < 2; ++j)
            acc[i][j] = (v8f){0.f, 0.f, 0.f, 0.f, 0.f, 0.f, 0.f, 0.f};

    uint4 rb[4];

    // ---- TDM: DMA a 128x64 bf16 weight tile (row stride Kw) into As[buf] ----
    // LDS offset = low 32 bits of the flat pointer (LDS aperture: addr[31:0]).
    auto tdm_loadA = [&](int buf, int s) {
        int kpos = s / spp;
        int kb   = (s - kpos * spp) << 6;
        u64 gaddr = (u64)(const void*)wA +
                    (((u64)(u32)mBlock * (u32)Kw + (u32)(kpos * Cin + kb)) << 1);
        u32 ldsA  = (u32)(u64)(const void*)&As[buf][0];
        u32x4 g0;
        g0[0] = 1u;                                   // count=1, user descriptor
        g0[1] = ldsA;                                 // lds_addr
        g0[2] = (u32)gaddr;                           // global_addr[31:0]
        g0[3] = ((u32)(gaddr >> 32) & 0x01FFFFFFu) | (2u << 30);  // addr[56:32] | type=2
        i32x8 g1;
        g1[0] = (int)((1u << 16) |                    // data_size = 2 bytes
                      (1u << 20) |                    // pad_enable
                      (4u << 22) |                    // pad_interval: 32 DWORDs (64 halves)
                      (3u << 25));                    // pad_amount: 4 DWORDs (8 halves)
        g1[1] = (int)(64u << 16);                     // tensor_dim0 = 64   (bits 79:48)
        g1[2] = (int)(128u << 16);                    // tensor_dim1 = 128  (bits 111:80)
        g1[3] = (int)(64u << 16);                     // tile_dim0 = 64     (bits 127:112)
        g1[4] = 128;                                  // tile_dim1 = 128    (bits 143:128)
        g1[5] = Kw;                                   // tensor_dim0_stride[31:0]
        g1[6] = 0;
        g1[7] = 0;
        i32x4 gz4 = (i32x4){0, 0, 0, 0};              // 2D tile: groups 2/3 unused
        i32x8 gz8 = (i32x8){0, 0, 0, 0, 0, 0, 0, 0};  // extra group (6-arg toolchain)
        __builtin_amdgcn_tensor_load_to_lds(g0, g1, gz4, gz4, gz8, 0);
    };

    auto gloadB = [&](int s) {
        int kpos = s / spp;
        int kb   = (s - kpos * spp) << 6;
        int dh   = kpos / ksize - hk;
        int dw   = kpos % ksize - hk;
        int hs = hB + dh, ws = wB + dw;
        bool valid = (hs >= 0) & (hs < HH) & (ws >= 0) & (ws < WW);
        const u16* bSrc = xIn + ((size_t)nimgB * HWSZ + hs * WW + ws) * Cin + kb + lch;
        uint4 z = make_uint4(0u, 0u, 0u, 0u);
        #pragma unroll
        for (int q = 0; q < 4; ++q)
            rb[q] = valid ? ((const uint4*)bSrc)[q] : z;
        if (valid) __builtin_prefetch((const char*)(bSrc + 64), 0, 1);
    };

    auto lstoreB = [&](int buf) {
        #pragma unroll
        for (int q = 0; q < 4; ++q)
            *(uint4*)&Bs[buf][lrow * LDR + lch + q * 8] = rb[q];
    };

    // prologue: stage step 0 (A via TDM, B via regs)
    if (tdmWave) tdm_loadA(0, 0);
    gloadB(0);
    lstoreB(0);
    if (tdmWave) __builtin_amdgcn_s_wait_tensorcnt(0);
    int cur = 0;

    for (int s = 0; s < S; ++s) {
        __syncthreads();                       // buf[cur] (TDM A + B stores) visible
        const bool more = (s + 1 < S);
        if (more) {
            if (tdmWave) tdm_loadA(cur ^ 1, s + 1);
            gloadB(s + 1);
        }

        // --- 16 WMMA on buffer `cur` (two 32-K sub-steps) ---
        #pragma unroll
        for (int k32 = 0; k32 < 64; k32 += 32) {
            Frag fa[4], fb[2];
            #pragma unroll
            for (int i = 0; i < 4; ++i) {
                int row = (wm0 + 16 * i + rr) * LDR + k32;
                fa[i].q[0] = *(const uint4*)&As[cur][row + 8 * half];       // K = 8h..
                fa[i].q[1] = *(const uint4*)&As[cur][row + 16 + 8 * half];  // K = 16+8h..
            }
            #pragma unroll
            for (int j = 0; j < 2; ++j) {
                int row = (wn0 + 16 * j + rr) * LDR + k32 + 8 * half;       // K = 8h.., 16+8h..
                fb[j].q[0] = *(const uint4*)&Bs[cur][row];
                fb[j].q[1] = *(const uint4*)&Bs[cur][row + 16];
            }
            #pragma unroll
            for (int i = 0; i < 4; ++i)
                #pragma unroll
                for (int j = 0; j < 2; ++j)
                    acc[i][j] = __builtin_amdgcn_wmma_f32_16x16x32_bf16(
                        false, fa[i].v, false, fb[j].v, (short)0, acc[i][j], false, false);
        }

        if (more) lstoreB(cur ^ 1);
        if (tdmWave) __builtin_amdgcn_s_wait_tensorcnt(0);   // A(s+1) landed in LDS
        cur ^= 1;
    }

    // ---- epilogue ----
    #pragma unroll
    for (int i = 0; i < 4; ++i) {
        #pragma unroll
        for (int j = 0; j < 2; ++j) {
            const int mg = mBlock + wm0 + 16 * i + 8 * half;   // channel base (v adds 0..7)
            const int pp = p0 + wn0 + 16 * j + rr;             // output pixel
            if constexpr (EPI == 0) {
                const int nimg = pp / HWSZ;
                const int hw   = pp - nimg * HWSZ;
                float* dst = outF + ((size_t)nimg * outCtot + chanBase + mg) * HWSZ + hw;
                #pragma unroll
                for (int v = 0; v < 8; ++v) {
                    float val = acc[i][j][v];
                    if (sc) val *= sc[mg + v];
                    if (bi) val += bi[mg + v];
                    dst[(size_t)v * HWSZ] = val;
                }
            } else {
                Pk8 pk;
                #pragma unroll
                for (int v = 0; v < 8; ++v) {
                    float val = acc[i][j][v];
                    if (sc) val *= sc[mg + v];
                    if (bi) val += bi[mg + v];
                    pk.s[v] = f2bf(val);
                }
                *(uint4*)&outBf[(size_t)pp * MOUT + mg] = pk.q;
            }
        }
    }
}

// ---------------------------------------------------------------------------
// 3x3 avg-pool (stride 1, pad 1, count_include_pad -> /9) + BN affine,
// u fp32 NCHW [16,256,56,56] -> allOut channels [768,1024).
// ---------------------------------------------------------------------------
__global__ void avgpool_kernel(const float* __restrict__ u, const float* __restrict__ s2,
                               const float* __restrict__ b2, float* __restrict__ allOut) {
    int idx = blockIdx.x * blockDim.x + threadIdx.x;
    if (idx >= NB * MOUT * HWSZ) return;
    int hw = idx % HWSZ;
    int c  = (idx / HWSZ) % MOUT;
    int n  = idx / (HWSZ * MOUT);
    int h = hw / WW, w = hw - h * WW;
    const float* base = u + ((size_t)n * MOUT + c) * HWSZ;
    float s = 0.f;
    #pragma unroll
    for (int dh = -1; dh <= 1; ++dh) {
        int hh = h + dh;
        if (hh < 0 || hh >= HH) continue;
        #pragma unroll
        for (int dw = -1; dw <= 1; ++dw) {
            int ww2 = w + dw;
            if (ww2 < 0 || ww2 >= WW) continue;
            s += base[hh * WW + ww2];
        }
    }
    float val = (s * (1.0f / 9.0f)) * s2[c] + b2[c];
    allOut[((size_t)n * 1024 + 768 + c) * HWSZ + hw] = val;
}

// ---------------------------------------------------------------------------
// Final sparse 1x1 select (float4-vectorized along W; 3136 % 4 == 0):
// out[n][oc] = allOut[n][ind[oc]] * sigmoid-scale[oc]
// ---------------------------------------------------------------------------
__global__ void gather_kernel(const float* __restrict__ allOut, const int* __restrict__ ind,
                              const float* __restrict__ scale, float* __restrict__ out) {
    int idx = blockIdx.x * blockDim.x + threadIdx.x;          // one float4 each
    if (idx >= NB * 512 * (HWSZ / 4)) return;
    int hw4 = (idx % (HWSZ / 4)) * 4;
    int oc  = (idx / (HWSZ / 4)) % 512;
    int n   = idx / ((HWSZ / 4) * 512);
    int s   = ind[oc];
    float k = scale[oc];
    float4 v = *(const float4*)&allOut[((size_t)n * 1024 + s) * HWSZ + hw4];
    v.x *= k; v.y *= k; v.z *= k; v.w *= k;
    *(float4*)&out[((size_t)n * 512 + oc) * HWSZ + hw4] = v;
}

// ---------------------------------------------------------------------------
extern "C" void kernel_launch(void* const* d_in, const int* in_sizes, int n_in,
                              void* d_out, int out_size, void* d_ws, size_t ws_size,
                              hipStream_t stream) {
    const float* x      = (const float*)d_in[0];
    const float* w_main = (const float*)d_in[1];
    const float* w_1x1  = (const float*)d_in[2];
    const float* w31    = (const float*)d_in[3];
    const float* bn31   = (const float*)d_in[4];
    const float* w33    = (const float*)d_in[5];
    const float* bn33   = (const float*)d_in[6];
    const float* wa1    = (const float*)d_in[7];
    const float* bna1   = (const float*)d_in[8];
    const float* bna2   = (const float*)d_in[9];
    const float* fw     = (const float*)d_in[10];
    const float* cscore = (const float*)d_in[11];
    float* out = (float*)d_out;

    // workspace carving (256B aligned)
    size_t off = 0;
    auto carve = [&](size_t bytes) -> char* {
        char* ptr = (char*)d_ws + off;
        off += (bytes + 255) & ~(size_t)255;
        return ptr;
    };
    u16*   xb     = (u16*)  carve((size_t)NPIX * CIN0 * 2);       // 51.4 MB
    u16*   tbuf   = (u16*)  carve((size_t)NPIX * MOUT * 2);       // 25.7 MB
    float* ubuf   = (float*)carve((size_t)NB * MOUT * HWSZ * 4);  // 51.4 MB
    float* allOut = (float*)carve((size_t)NB * 1024 * HWSZ * 4);  // 205.5 MB
    u16*   wbMain = (u16*)  carve((size_t)MOUT * CIN0 * 9 * 2);
    u16*   wb33   = (u16*)  carve((size_t)MOUT * MOUT * 9 * 2);
    u16*   wb1    = (u16*)  carve((size_t)MOUT * CIN0 * 2);
    u16*   wb31   = (u16*)  carve((size_t)MOUT * CIN0 * 2);
    u16*   wba1   = (u16*)  carve((size_t)MOUT * CIN0 * 2);
    float* bnbuf  = (float*)carve(2048 * 4);
    int*   ind    = (int*)  carve(512 * 4);
    float* scl    = (float*)carve(512 * 4);

    const float* s31 = bnbuf;            const float* b31 = bnbuf + 256;
    const float* s33 = bnbuf + 512;      const float* b33 = bnbuf + 768;
    const float* sa1 = bnbuf + 1024;     const float* ba1 = bnbuf + 1280;
    const float* s2  = bnbuf + 1536;     const float* b2  = bnbuf + 1792;

    // 1) small preps
    bnprep_kernel<<<1, 256, 0, stream>>>(bn31, bn33, bna1, bna2, bnbuf);
    topk_kernel<<<1, 1024, 0, stream>>>(cscore, ind, scl);

    // 2) activation transpose to bf16 NHWC
    xpose_kernel<<<dim3(NPIX / 32, CIN0 / 32), 256, 0, stream>>>(x, xb);

    // 3) weight packing (fold fuse_weight + BN scale)
    pack3_kernel<<<(MOUT * CIN0 * 9 + 255) / 256, 256, 0, stream>>>(w_main, wbMain, CIN0, fw, 0);
    pack3_kernel<<<(MOUT * MOUT * 9 + 255) / 256, 256, 0, stream>>>(w33, wb33, MOUT, fw, -1);
    pack1_kernel<<<(MOUT * CIN0 + 255) / 256, 256, 0, stream>>>(w_1x1, wb1, CIN0, fw, 1, nullptr);
    pack1_kernel<<<(MOUT * CIN0 + 255) / 256, 256, 0, stream>>>(w31, wb31, CIN0, fw, 2, s31);
    pack1_kernel<<<(MOUT * CIN0 + 255) / 256, 256, 0, stream>>>(wa1, wba1, CIN0, fw, 3, sa1);

    // 4) WMMA convs: grid = 392 pixel tiles x 2 channel tiles
    dim3 cg(NPIX / 128, 2);
    // branch 0: 3x3 main -> allOut[0:256)
    conv_wmma_kernel<0><<<cg, 256, 0, stream>>>(wbMain, xb, CIN0, 3,
                                                nullptr, nullptr, allOut, 0, 1024, nullptr);
    // branch 1: 1x1 -> allOut[256:512)
    conv_wmma_kernel<0><<<cg, 256, 0, stream>>>(wb1, xb, CIN0, 1,
                                                nullptr, nullptr, allOut, 256, 1024, nullptr);
    // branch 2a: 1x1 + BN bias -> bf16 NHWC intermediate t
    conv_wmma_kernel<1><<<cg, 256, 0, stream>>>(wb31, xb, CIN0, 1,
                                                nullptr, b31, nullptr, 0, 0, tbuf);
    // branch 2b: 3x3 on t + BN affine -> allOut[512:768)
    conv_wmma_kernel<0><<<cg, 256, 0, stream>>>(wb33, tbuf, MOUT, 3,
                                                s33, b33, allOut, 512, 1024, nullptr);
    // branch 3a: 1x1 + BN(bna1) -> u (fp32 NCHW)
    conv_wmma_kernel<0><<<cg, 256, 0, stream>>>(wba1, xb, CIN0, 1,
                                                nullptr, ba1, ubuf, 0, MOUT, nullptr);

    // 5) avg-pool + BN(bna2) -> allOut[768:1024)
    avgpool_kernel<<<(NB * MOUT * HWSZ + 255) / 256, 256, 0, stream>>>(ubuf, s2, b2, allOut);

    // 6) top-k gather + sigmoid scale -> d_out
    gather_kernel<<<(NB * 512 * (HWSZ / 4) + 255) / 256, 256, 0, stream>>>(allOut, ind, scl, out);
}
